// ST_GCN_non_SF_90305982366370
// MI455X (gfx1250) — matrix operand order repaired and checked by reading, not compiled, not run on hardware
//
#include <hip/hip_runtime.h>
#include <hip/hip_bf16.h>

// ---------------------------------------------------------------------------
// Types for CDNA5 WMMA (wave32, v_wmma_f32_16x16x32_bf16)
// ---------------------------------------------------------------------------
typedef __attribute__((ext_vector_type(16))) __bf16 v16bf;
typedef __attribute__((ext_vector_type(8)))  float  v8f;

union FragBF16 {
    v16bf v;
    uint4 q[2];   // 32 bytes
};

// f32 -> bf16 (round-to-nearest, ties away): cheap 2-op conversion.
__device__ __forceinline__ unsigned pack2_bf16(float lo, float hi) {
    union { float f; unsigned u; } a, b;
    a.f = lo; b.f = hi;
    return ((a.u + 0x8000u) >> 16) | ((b.u + 0x8000u) & 0xFFFF0000u);
}

// ---------------------------------------------------------------------------
// Generic GEMM: C[M,N] = A[M,K] * B[K,N] (+bias, +optional ReLU)
// f32 in global memory, bf16 WMMA math, f32 accumulate.
// Block: 256 threads = 8 waves. Block tile BM=64 x BN=64; waves arranged
// 4 (rows) x 2 (cols); each wave owns a 16x32 D tile (two f32 accumulators),
// reusing its A fragment across two WMMAs. K staged in BK=64 chunks
// -> 4 WMMA per stage. Staging is branchless & fully unrolled; a K-tail
// gets one guarded stage.
// REQUIRES: M % BM == 0 (true for all call sites here).
// ---------------------------------------------------------------------------
#define BM 64
#define BN 64
#define BK 64

__global__ __launch_bounds__(256)
void gemm_bf16_wmma(const float* __restrict__ A, const float* __restrict__ B,
                    const float* __restrict__ bias, float* __restrict__ C,
                    int M, int N, int K, int relu)
{
    __shared__ alignas(16) unsigned short sA[BM * BK];    // 8 KB
    __shared__ alignas(16) unsigned short sBt[BN * BK];   // 8 KB (B transposed)

    const int tid     = threadIdx.x;
    const int lane    = tid & 31;
    const int wave    = tid >> 5;
    const int waveRow = wave >> 1;   // 0..3  (16-row tile)
    const int waveCol = wave & 1;    // 0..1  (32-col tile)
    const int rowBase = blockIdx.y * BM;
    const int colBase = blockIdx.x * BN;

    const int half = lane >> 4;      // lane half selects K sub-blocks
    const int mr   = lane & 15;      // A-row / D-col within 16x16 tile

    unsigned* sA32  = (unsigned*)sA;
    unsigned* sBt32 = (unsigned*)sBt;

    v8f acc0 = {0.f, 0.f, 0.f, 0.f, 0.f, 0.f, 0.f, 0.f};
    v8f acc1 = {0.f, 0.f, 0.f, 0.f, 0.f, 0.f, 0.f, 0.f};

    for (int k0 = 0; k0 < K; k0 += BK) {
        const bool full = (k0 + BK) <= K;
        if (full) {
            // ---- A tile: BM*BK/2 = 2048 packed pairs, 8 per thread ----
            #pragma unroll
            for (int it = 0; it < (BM * BK / 2) / 256; ++it) {
                int p = tid + it * 256;
                int r = p >> 5;                 // BK/2 = 32 pairs per row
                int c = (p & 31) << 1;
                const float* ap = &A[(long long)(rowBase + r) * K + k0 + c];
                sA32[p] = pack2_bf16(ap[0], ap[1]);
            }
            // ---- B tile transposed: BN*BK/2 = 2048 pairs, 8 per thread ----
            #pragma unroll
            for (int it = 0; it < (BN * BK / 2) / 256; ++it) {
                int p = tid + it * 256;
                int n = p >> 5;
                int c = (p & 31) << 1;
                int gn  = colBase + n;
                int gnc = gn < N ? gn : N - 1;        // branchless N guard
                float m = gn < N ? 1.f : 0.f;
                const float* bp = &B[(long long)(k0 + c) * N + gnc];
                sBt32[p] = pack2_bf16(bp[0] * m, bp[N] * m);
            }
        } else {
            // ---- guarded tail stage (K % BK != 0), branchless clamps ----
            #pragma unroll
            for (int it = 0; it < (BM * BK / 2) / 256; ++it) {
                int p = tid + it * 256;
                int r = p >> 5;
                int c = (p & 31) << 1;
                int gc0 = k0 + c, gc1 = k0 + c + 1;
                int c0 = gc0 < K ? gc0 : K - 1;
                int c1 = gc1 < K ? gc1 : K - 1;
                float m0 = gc0 < K ? 1.f : 0.f;
                float m1 = gc1 < K ? 1.f : 0.f;
                const float* ar = &A[(long long)(rowBase + r) * K];
                sA32[p] = pack2_bf16(ar[c0] * m0, ar[c1] * m1);
            }
            #pragma unroll
            for (int it = 0; it < (BN * BK / 2) / 256; ++it) {
                int p = tid + it * 256;
                int n = p >> 5;
                int c = (p & 31) << 1;
                int gn  = colBase + n;
                int gnc = gn < N ? gn : N - 1;
                int gk0 = k0 + c, gk1 = k0 + c + 1;
                int k0c = gk0 < K ? gk0 : K - 1;
                int k1c = gk1 < K ? gk1 : K - 1;
                float m0 = (gn < N && gk0 < K) ? 1.f : 0.f;
                float m1 = (gn < N && gk1 < K) ? 1.f : 0.f;
                sBt32[p] = pack2_bf16(B[(long long)k0c * N + gnc] * m0,
                                      B[(long long)k1c * N + gnc] * m1);
            }
        }
        __syncthreads();

        // ---- fragments in contiguous k order (k = 16h + s), same for A and B; 4 WMMA per stage ----
        const uint4* pa = (const uint4*)sA;    // 8 ushorts per uint4
        const uint4* pb = (const uint4*)sBt;
        const int arow  = waveRow * 16 + mr;           // A lane row
        const int brow0 = waveCol * 32 + mr;           // Bt lane row, subtile 0
        const int brow1 = waveCol * 32 + 16 + mr;      // Bt lane row, subtile 1
        #pragma unroll
        for (int kc = 0; kc < 2; ++kc) {
            FragBF16 a, b0, b1;
            // A lane (half h): K = kc*32 + h*16 + 0..15 (contiguous in sA row)
            a.q[0] = pa[arow * 8 + kc * 4 + half * 2];
            a.q[1] = pa[arow * 8 + kc * 4 + half * 2 + 1];
            // B lane (half h): K = kc*32 + h*16 + 0..15 (contiguous in sBt row)
            b0.q[0] = pb[brow0 * 8 + kc * 4 + half * 2];
            b0.q[1] = pb[brow0 * 8 + kc * 4 + half * 2 + 1];
            b1.q[0] = pb[brow1 * 8 + kc * 4 + half * 2];
            b1.q[1] = pb[brow1 * 8 + kc * 4 + half * 2 + 1];
            acc0 = __builtin_amdgcn_wmma_f32_16x16x32_bf16(
                false, a.v, false, b0.v, (short)0, acc0, false, false);
            acc1 = __builtin_amdgcn_wmma_f32_16x16x32_bf16(
                false, a.v, false, b1.v, (short)0, acc1, false, false);
        }
        __syncthreads();
    }

    // ---- store D: lane col = mr, VGPR r -> row r + half*8 ----
    #pragma unroll
    for (int c = 0; c < 2; ++c) {
        int col = colBase + waveCol * 32 + c * 16 + mr;
        if (col < N) {
            const v8f& acc = c ? acc1 : acc0;
            float bv = bias ? bias[col] : 0.f;
            #pragma unroll
            for (int r = 0; r < 8; ++r) {
                int row = rowBase + waveRow * 16 + r + half * 8;
                float v = acc[r] + bv;
                if (relu) v = fmaxf(v, 0.f);
                C[(long long)row * N + col] = v;
            }
        }
    }
}

// ---------------------------------------------------------------------------
// Graph kernels (F is a power of two; shifts passed from host)
// ---------------------------------------------------------------------------
__global__ __launch_bounds__(256)
void deg_kernel(const long long* __restrict__ tgt, float* __restrict__ deg, int E)
{
    int i = blockIdx.x * blockDim.x + threadIdx.x;
    if (i < E) atomicAdd(&deg[(int)tgt[i]], 1.0f);
}

__global__ __launch_bounds__(256)
void dinv_kernel(const float* __restrict__ deg, float* __restrict__ dinv, int N)
{
    int i = blockIdx.x * blockDim.x + threadIdx.x;
    if (i < N) dinv[i] = rsqrtf(deg[i] + 1.0f);   // +1 self-loop; always > 0
}

// h[i,f] = dinv[i]^2 * xw[i,f] + bias[f]   (self-loop message + bias)
__global__ __launch_bounds__(256)
void selfloop_bias_kernel(const float* __restrict__ xw, const float* __restrict__ dinv,
                          const float* __restrict__ bias, float* __restrict__ h,
                          int N, int logF)
{
    int i = blockIdx.x * blockDim.x + threadIdx.x;
    int total = N << logF;
    if (i < total) {
        int node = i >> logF;
        int f    = i & ((1 << logF) - 1);
        float d  = dinv[node];
        h[i] = d * d * xw[i] + bias[f];
    }
}

// edge scatter: h[tgt] += dinv[src]*dinv[tgt] * xw[src]; float4 per thread
__global__ __launch_bounds__(256)
void scatter_kernel(const long long* __restrict__ src, const long long* __restrict__ tgt,
                    const float* __restrict__ dinv, const float* __restrict__ xw,
                    float* __restrict__ h, int E, int logFQ)
{
    int idx = blockIdx.x * blockDim.x + threadIdx.x;
    int total = E << logFQ;
    if (idx >= total) return;
    int e = idx >> logFQ;
    int j = (idx & ((1 << logFQ) - 1)) << 2;
    int F = 1 << (logFQ + 2);
    int s = (int)src[e];
    int t = (int)tgt[e];
    float nrm = dinv[s] * dinv[t];
    const float4 v = *(const float4*)&xw[(long long)s * F + j];
    float* dst = &h[(long long)t * F + j];
    atomicAdd(dst + 0, v.x * nrm);
    atomicAdd(dst + 1, v.y * nrm);
    atomicAdd(dst + 2, v.z * nrm);
    atomicAdd(dst + 3, v.w * nrm);
}

__global__ __launch_bounds__(256)
void relu_kernel(float* __restrict__ h, int total)
{
    int i = blockIdx.x * blockDim.x + threadIdx.x;
    if (i < total) h[i] = fmaxf(h[i], 0.f);
}

// global sum pool: g[batch[node], f] += h[node, f]; float4 per thread
__global__ __launch_bounds__(256)
void pool_kernel(const long long* __restrict__ batch, const float* __restrict__ h,
                 float* __restrict__ g, int N, int logFQ)
{
    int idx = blockIdx.x * blockDim.x + threadIdx.x;
    int total = N << logFQ;
    if (idx >= total) return;
    int node = idx >> logFQ;
    int j = (idx & ((1 << logFQ) - 1)) << 2;
    int F = 1 << (logFQ + 2);
    int b = (int)batch[node];
    const float4 v = *(const float4*)&h[(long long)node * F + j];
    float* dst = &g[(long long)b * F + j];
    atomicAdd(dst + 0, v.x);
    atomicAdd(dst + 1, v.y);
    atomicAdd(dst + 2, v.z);
    atomicAdd(dst + 3, v.w);
}

// final layer: out[i] = g4[i,:] . w + b  (K small, N=1)
__global__ __launch_bounds__(256)
void final_dot_kernel(const float* __restrict__ g4, const float* __restrict__ w,
                      const float* __restrict__ b, float* __restrict__ out,
                      int G, int K)
{
    int i = blockIdx.x * blockDim.x + threadIdx.x;
    if (i >= G) return;
    float s = b[0];
    for (int k = 0; k < K; ++k) s += g4[(long long)i * K + k] * w[k];
    out[i] = s;
}

// ---------------------------------------------------------------------------
// Host launch
// ---------------------------------------------------------------------------
static inline int ceil_div(int a, int b) { return (a + b - 1) / b; }
static inline int ilog2(int v) { int l = 0; while ((1 << l) < v) ++l; return l; }

static void launch_gemm(const float* A, const float* B, const float* bias,
                        float* C, int M, int N, int K, int relu, hipStream_t s)
{
    dim3 grid(ceil_div(N, BN), ceil_div(M, BM));
    gemm_bf16_wmma<<<grid, 256, 0, s>>>(A, B, bias, C, M, N, K, relu);
}

extern "C" void kernel_launch(void* const* d_in, const int* in_sizes, int n_in,
                              void* d_out, int out_size, void* d_ws, size_t ws_size,
                              hipStream_t stream)
{
    (void)n_in; (void)ws_size; (void)out_size;
    const float*     x     = (const float*)d_in[0];
    const long long* ei    = (const long long*)d_in[1];   // [2,E] int64
    const long long* batch = (const long long*)d_in[2];   // [N]   int64
    // d_in[3] = num_graphs scalar on device (G fixed at 2048 per setup)
    const float* W1  = (const float*)d_in[4];
    const float* b1  = (const float*)d_in[5];
    const float* W2  = (const float*)d_in[6];
    const float* b2  = (const float*)d_in[7];
    const float* l1w = (const float*)d_in[8];
    const float* l1b = (const float*)d_in[9];
    const float* l2w = (const float*)d_in[10];
    const float* l2b = (const float*)d_in[11];
    const float* l3w = (const float*)d_in[12];
    const float* l3b = (const float*)d_in[13];
    const float* l4w = (const float*)d_in[14];
    const float* l4b = (const float*)d_in[15];
    const float* l5w = (const float*)d_in[16];
    const float* l5b = (const float*)d_in[17];

    const int N    = in_sizes[2];             // 131072 nodes
    const int E    = in_sizes[1] / 2;         // 1048576 edges
    const int HID  = in_sizes[5];             // 128
    const int INC  = in_sizes[4] / HID;       // 131
    const int OUTC = in_sizes[7];             // 64
    const int D1   = in_sizes[9];             // 2048
    const int D2   = in_sizes[11];            // 1024
    const int D3   = in_sizes[13];            // 128
    const int D4   = in_sizes[15];            // 32
    const int G    = 2048;                    // num_graphs (fixed by setup)

    const long long* src = ei;
    const long long* tgt = ei + E;

    // ---- workspace layout (floats), with buffer reuse ----
    float* ws   = (float*)d_ws;
    float* deg  = ws;                                  // N
    float* dinv = deg + N;                             // N
    float* bufA = dinv + N;                            // N*HID (xw1, later xw2)
    float* bufB = bufA + (size_t)N * HID;              // N*HID (h1, later h2)
    float* g    = bufB + (size_t)N * HID;              // G*OUTC
    float* g1   = g  + (size_t)G * OUTC;               // G*D1
    float* g2   = g1 + (size_t)G * D1;                 // G*D2
    float* g3   = g2 + (size_t)G * D2;                 // G*D3
    float* g4   = g3 + (size_t)G * D3;                 // G*D4

    // ---- degree + normalization ----
    hipMemsetAsync(deg, 0, (size_t)N * sizeof(float), stream);
    deg_kernel<<<ceil_div(E, 256), 256, 0, stream>>>(tgt, deg, E);
    dinv_kernel<<<ceil_div(N, 256), 256, 0, stream>>>(deg, dinv, N);

    // ---- GCN layer 1: xw1 = x @ W1 ; h1 = Agg(xw1) + b1 ; ReLU ----
    launch_gemm(x, W1, nullptr, bufA, N, HID, INC, 0, stream);
    selfloop_bias_kernel<<<ceil_div(N * HID, 256), 256, 0, stream>>>(
        bufA, dinv, b1, bufB, N, ilog2(HID));
    scatter_kernel<<<ceil_div(E * (HID / 4), 256), 256, 0, stream>>>(
        src, tgt, dinv, bufA, bufB, E, ilog2(HID / 4));
    relu_kernel<<<ceil_div(N * HID, 256), 256, 0, stream>>>(bufB, N * HID);

    // ---- GCN layer 2: xw2 = h1 @ W2 ; h2 = Agg(xw2) + b2 ----
    launch_gemm(bufB, W2, nullptr, bufA, N, OUTC, HID, 0, stream);
    selfloop_bias_kernel<<<ceil_div(N * OUTC, 256), 256, 0, stream>>>(
        bufA, dinv, b2, bufB, N, ilog2(OUTC));
    scatter_kernel<<<ceil_div(E * (OUTC / 4), 256), 256, 0, stream>>>(
        src, tgt, dinv, bufA, bufB, E, ilog2(OUTC / 4));

    // ---- global sum pool ----
    hipMemsetAsync(g, 0, (size_t)G * OUTC * sizeof(float), stream);
    pool_kernel<<<ceil_div(N * (OUTC / 4), 256), 256, 0, stream>>>(
        batch, bufB, g, N, ilog2(OUTC / 4));

    // ---- MLP head (bias+ReLU fused into WMMA GEMMs) ----
    launch_gemm(g,  l1w, l1b, g1, G, D1, OUTC, 1, stream);
    launch_gemm(g1, l2w, l2b, g2, G, D2, D1,   1, stream);
    launch_gemm(g2, l3w, l3b, g3, G, D3, D2,   1, stream);
    launch_gemm(g3, l4w, l4b, g4, G, D4, D3,   1, stream);
    final_dot_kernel<<<ceil_div(G, 256), 256, 0, stream>>>(
        g4, l5w, l5b, (float*)d_out, G, D4);
}
